// MultiHeadSelfAttention_70763881168987
// MI455X (gfx1250) — matrix operand order repaired and checked by reading, not compiled, not run on hardware
//
#include <hip/hip_runtime.h>

// ---------------------------------------------------------------------------
// MHSA for gfx1250 (MI455X): bf16 WMMA flash attention + TDM async staging.
// B=4, T=2048, D=512, H=8, DH=64
// ---------------------------------------------------------------------------

#define B_  4
#define T_  2048
#define D_  512
#define H_  8
#define DH_ 64

typedef __attribute__((ext_vector_type(16))) __bf16 v16bf;
typedef __attribute__((ext_vector_type(8)))  __bf16 v8bf;
typedef __attribute__((ext_vector_type(8)))  float  v8f;
typedef __attribute__((ext_vector_type(4)))  unsigned int u32x4;
typedef __attribute__((ext_vector_type(8)))  int i32x8;
typedef __attribute__((ext_vector_type(4)))  int i32x4;

static __device__ __forceinline__ v8f wmma_bf16(v16bf a, v16bf b, v8f c) {
  // (neg_a, A, neg_b, B, c_mod, C, reuse_a, reuse_b)
  return __builtin_amdgcn_wmma_f32_16x16x32_bf16(false, a, false, b, (short)0, c,
                                                 false, false);
}

static __device__ __forceinline__ v16bf concat8(v8bf lo, v8bf hi) {
  v16bf r;
#pragma unroll
  for (int i = 0; i < 8; ++i) { r[i] = lo[i]; r[i + 8] = hi[i]; }
  return r;
}

// ---------------------------------------------------------------------------
// TDM: async 2D tile load Global -> LDS (bf16 elements, data_size code 1).
// D# built per CDNA5 ISA ch.8: group0 = {count|flags, lds_addr, ga_lo,
// ga_hi|type<<30}; group1 packs data_size, tensor dims, tile dims, stride0.
// Dims/strides in element units; addresses in bytes.
// ---------------------------------------------------------------------------
static __device__ __forceinline__ void tdm_load_2d_bf16(
    unsigned lds_off, const void* gaddr,
    unsigned tensor_d0, unsigned tensor_d1, unsigned stride0,
    unsigned tile_d0, unsigned tile_d1) {
  unsigned long long ga = (unsigned long long)gaddr;
  u32x4 g0;
  g0[0] = 1u;                                         // count=1, user-mode
  g0[1] = lds_off;                                    // lds_addr (bytes)
  g0[2] = (unsigned)(ga & 0xFFFFFFFFu);               // global_addr[31:0]
  g0[3] = (unsigned)((ga >> 32) & 0x01FFFFFFu) | (2u << 30);  // [56:32]|type=2
  i32x8 g1;
  g1[0] = (int)(1u << 16);                            // data_size=1 -> 2 bytes
  g1[1] = (int)(tensor_d0 << 16);                     // tensor_dim0[15:0]
  g1[2] = (int)((tensor_d0 >> 16) | (tensor_d1 << 16));
  g1[3] = (int)((tensor_d1 >> 16) | (tile_d0 << 16)); // tile_dim0
  g1[4] = (int)(tile_d1 & 0xFFFFu);                   // tile_dim1 (tile_dim2=0)
  g1[5] = (int)stride0;                               // tensor_dim0_stride lo
  g1[6] = 0;
  g1[7] = 0;
  i32x4 z4 = {0, 0, 0, 0};
#if __has_include(<hip/amd_detail/amd_gfx1250_TDM.h>)
  i32x8 z8 = {0, 0, 0, 0, 0, 0, 0, 0};
  __builtin_amdgcn_tensor_load_to_lds(g0, g1, z4, z4, z8, 0);
#else
  __builtin_amdgcn_tensor_load_to_lds(g0, g1, z4, z4, 0);
#endif
}

// ---------------------------------------------------------------------------
// 0) Convert + transpose weights: W[K][N] f32 -> WT[N][K] bf16
// ---------------------------------------------------------------------------
__global__ void cvt_transpose_kernel(const float* __restrict__ W,
                                     __bf16* __restrict__ WT, int K, int N) {
  int idx = blockIdx.x * 256 + threadIdx.x;
  if (idx < K * N) {
    int n = idx / K;
    int k = idx - n * K;
    WT[idx] = (__bf16)W[(size_t)k * N + n];
  }
}

// ---------------------------------------------------------------------------
// 1) QKV projection: one wave computes a 16x128 tile of x@Wqkv.
//    Q,K -> [B,H,T,DH] bf16; V -> [B,H,DH,T] bf16 (transposed for PV gemm).
// ---------------------------------------------------------------------------
__global__ void qkv_kernel(const float* __restrict__ x,
                           const __bf16* __restrict__ WT,  // [3D][D] bf16
                           __bf16* __restrict__ Qb, __bf16* __restrict__ Kb,
                           __bf16* __restrict__ Vt) {
  const int lane = threadIdx.x;
  const int ln = lane & 15, hf = lane >> 4;
  const int mbase = blockIdx.x * 16;   // row tile in flattened [B*T]
  const int nbase = blockIdx.y * 128;  // col tile in [3D]

  v8f acc[8] = {};
  const float* xrow = x + (size_t)(mbase + ln) * D_;

  for (int kk = 0; kk < D_; kk += 32) {
    v8f xlo = *(const v8f*)(xrow + kk + hf * 16);
    v8f xhi = *(const v8f*)(xrow + kk + 8 + hf * 16);
    v16bf A;
#pragma unroll
    for (int i = 0; i < 8; ++i) { A[i] = (__bf16)xlo[i]; A[i + 8] = (__bf16)xhi[i]; }
#pragma unroll
    for (int nt = 0; nt < 8; ++nt) {
      const __bf16* wrow = WT + (size_t)(nbase + nt * 16 + ln) * D_;
      v16bf Bm = *(const v16bf*)(wrow + kk + hf * 16);   // 32B contiguous
      acc[nt] = wmma_bf16(A, Bm, acc[nt]);
    }
  }

  const int b = mbase / T_;  // 16 | T -> whole tile in one batch
#pragma unroll
  for (int nt = 0; nt < 8; ++nt) {
    const int n = nbase + nt * 16 + ln;
    const int seg = n >> 9;        // 0=Q 1=K 2=V
    const int nn = n & 511;
    const int h = nn >> 6, d = nn & 63;
    const size_t bh = (size_t)(b * H_ + h);
#pragma unroll
    for (int r = 0; r < 8; ++r) {
      const int tok = (mbase + r + 8 * hf) & (T_ - 1);
      const __bf16 val = (__bf16)acc[nt][r];
      if (seg == 0)      Qb[(bh * T_ + tok) * DH_ + d] = val;
      else if (seg == 1) Kb[(bh * T_ + tok) * DH_ + d] = val;
      else               Vt[(bh * DH_ + d) * T_ + tok] = val;   // transposed
    }
  }
}

// ---------------------------------------------------------------------------
// 2) Flash attention: one wave per (b, h, 32-row q tile); 32-key tiles with
//    TDM double-buffered K/V staging in LDS. 16 WMMA per key tile.
// ---------------------------------------------------------------------------
__global__ void attn_kernel(const __bf16* __restrict__ Qb,
                            const __bf16* __restrict__ Kb,
                            const __bf16* __restrict__ Vt,
                            const float* __restrict__ bias,
                            const int* __restrict__ mask,
                            __bf16* __restrict__ attn_out) {
  __shared__ __bf16 Kl[2][32 * DH_];   // [buf][32 tokens][64 d]
  __shared__ __bf16 Vl[2][DH_ * 32];   // [buf][64 d][32 tokens]
  __shared__ __bf16 Pl[32 * 32];       // P staging: D-layout -> A-layout

  const int lane = threadIdx.x;
  const int ln = lane & 15, hf = lane >> 4;
  const int h = blockIdx.y, b = blockIdx.z;
  const int bh = b * H_ + h;
  const int qbase = blockIdx.x * 32;
  const float scale = 0.125f;          // 1/sqrt(DH)
  const float NEG_INF = -__builtin_inff();

  const __bf16* kptr = Kb + (size_t)bh * T_ * DH_;
  const __bf16* vptr = Vt + (size_t)bh * DH_ * T_;

  // Q fragments (A-layout) for two 16-row blocks; invariant over key loop.
  v16bf Aq[2][2];
#pragma unroll
  for (int rb = 0; rb < 2; ++rb) {
    const __bf16* qrow = Qb + ((size_t)bh * T_ + qbase + rb * 16 + ln) * DH_;
    Aq[rb][0] = concat8(*(const v8bf*)(qrow + hf * 16),
                        *(const v8bf*)(qrow + 8 + hf * 16));
    Aq[rb][1] = concat8(*(const v8bf*)(qrow + 32 + hf * 16),
                        *(const v8bf*)(qrow + 40 + hf * 16));
  }

  v8f O[2][4] = {};                    // [row-block][d-tile]
  float mrow[2][8], lrow[2][8];
#pragma unroll
  for (int rb = 0; rb < 2; ++rb)
#pragma unroll
    for (int r = 0; r < 8; ++r) { mrow[rb][r] = NEG_INF; lrow[rb][r] = 0.f; }

  // Prologue: stage key tile 0 into buffer 0.
  tdm_load_2d_bf16((unsigned)(size_t)&Kl[0][0], kptr,        DH_, T_, DH_, DH_, 32);
  tdm_load_2d_bf16((unsigned)(size_t)&Vl[0][0], vptr,        T_, DH_, T_,  32, DH_);

  for (int kb = 0; kb < T_; kb += 32) {
    const int cur = (kb >> 5) & 1;
    if (kb + 32 < T_) {
      const int nxt = cur ^ 1;
      tdm_load_2d_bf16((unsigned)(size_t)&Kl[nxt][0],
                       kptr + (size_t)(kb + 32) * DH_, DH_, T_, DH_, DH_, 32);
      tdm_load_2d_bf16((unsigned)(size_t)&Vl[nxt][0],
                       vptr + (kb + 32), T_, DH_, T_, 32, DH_);
      __builtin_amdgcn_s_wait_tensorcnt((short)2);  // current pair landed
    } else {
      __builtin_amdgcn_s_wait_tensorcnt((short)0);
    }
    asm volatile("" ::: "memory");     // don't hoist LDS reads above the wait

    // K B-fragments (shared by both row-blocks): d = w*32 + hf*16 + j.
    const __bf16* k0 = &Kl[cur][(size_t)(0 + ln) * DH_];
    const __bf16* k1 = &Kl[cur][(size_t)(16 + ln) * DH_];
    const v16bf Bk00 = *(const v16bf*)(k0 + hf * 16);
    const v16bf Bk01 = *(const v16bf*)(k0 + 32 + hf * 16);
    const v16bf Bk10 = *(const v16bf*)(k1 + hf * 16);
    const v16bf Bk11 = *(const v16bf*)(k1 + 32 + hf * 16);

    const int kc0 = kb + ln, kc1 = kb + 16 + ln;
    const float msk0 = mask[b * T_ + kc0] ? 0.f : NEG_INF;
    const float msk1 = mask[b * T_ + kc1] ? 0.f : NEG_INF;

    __syncthreads();                   // P readers of prev iter are done
#pragma unroll
    for (int rb = 0; rb < 2; ++rb) {
      v8f S0 = {}, S1 = {};
      S0 = wmma_bf16(Aq[rb][0], Bk00, S0);
      S0 = wmma_bf16(Aq[rb][1], Bk01, S0);
      S1 = wmma_bf16(Aq[rb][0], Bk10, S1);
      S1 = wmma_bf16(Aq[rb][1], Bk11, S1);

      const float* br0 = bias + ((size_t)b * T_ + qbase + rb * 16) * T_;
      float alpha[8];
#pragma unroll
      for (int r = 0; r < 8; ++r) {
        // D layout: reg r of this lane = row (r + 8*hf), column ln.
        const float* brr = br0 + (size_t)(r + 8 * hf) * T_;
        float s0 = fmaf(S0[r], scale, brr[kc0]) + msk0;
        float s1 = fmaf(S1[r], scale, brr[kc1]) + msk1;
        float t = fmaxf(s0, s1);       // row reduce: 16-lane butterfly
        t = fmaxf(t, __shfl_xor(t, 1));
        t = fmaxf(t, __shfl_xor(t, 2));
        t = fmaxf(t, __shfl_xor(t, 4));
        t = fmaxf(t, __shfl_xor(t, 8));
        const float mn = fmaxf(mrow[rb][r], t);
        alpha[r] = __expf(mrow[rb][r] - mn);
        mrow[rb][r] = mn;
        const float p0 = __expf(s0 - mn);
        const float p1 = __expf(s1 - mn);
        float rs = p0 + p1;
        rs += __shfl_xor(rs, 1);
        rs += __shfl_xor(rs, 2);
        rs += __shfl_xor(rs, 4);
        rs += __shfl_xor(rs, 8);
        lrow[rb][r] = lrow[rb][r] * alpha[r] + rs;
        Pl[(rb * 16 + r + 8 * hf) * 32 + ln]      = (__bf16)p0;
        Pl[(rb * 16 + r + 8 * hf) * 32 + 16 + ln] = (__bf16)p1;
      }
#pragma unroll
      for (int r = 0; r < 8; ++r) {
        O[rb][0][r] *= alpha[r]; O[rb][1][r] *= alpha[r];
        O[rb][2][r] *= alpha[r]; O[rb][3][r] *= alpha[r];
      }
    }
    __syncthreads();

    // PV: V B-fragments shared by both row-blocks.
    v16bf Bv[4];
#pragma unroll
    for (int dt = 0; dt < 4; ++dt)
      Bv[dt] = *(const v16bf*)(&Vl[cur][(size_t)(dt * 16 + ln) * 32 + hf * 16]);
#pragma unroll
    for (int rb = 0; rb < 2; ++rb) {
      const v16bf Ap = concat8(
          *(const v8bf*)(&Pl[(rb * 16 + ln) * 32 + hf * 16]),
          *(const v8bf*)(&Pl[(rb * 16 + ln) * 32 + 8 + hf * 16]));
#pragma unroll
      for (int dt = 0; dt < 4; ++dt)
        O[rb][dt] = wmma_bf16(Ap, Bv[dt], O[rb][dt]);
    }
  }

  // Finalize: divide by softmax denominator, write bf16 [B,T,D].
#pragma unroll
  for (int rb = 0; rb < 2; ++rb) {
#pragma unroll
    for (int r = 0; r < 8; ++r) {
      const float inv = 1.0f / lrow[rb][r];
      const int tok = qbase + rb * 16 + r + 8 * hf;
      __bf16* orow = attn_out + ((size_t)b * T_ + tok) * D_ + h * DH_;
#pragma unroll
      for (int dt = 0; dt < 4; ++dt)
        orow[dt * 16 + ln] = (__bf16)(O[rb][dt][r] * inv);
    }
  }
}

// ---------------------------------------------------------------------------
// 3) Output projection: out[8192 x 512] f32 = attn_bf16 * WoutT (16x128 tile)
// ---------------------------------------------------------------------------
__global__ void proj_kernel(const __bf16* __restrict__ A,
                            const __bf16* __restrict__ WT,  // [D][D] bf16
                            float* __restrict__ out) {
  const int lane = threadIdx.x;
  const int ln = lane & 15, hf = lane >> 4;
  const int mbase = blockIdx.x * 16;
  const int nbase = blockIdx.y * 128;

  v8f acc[8] = {};
  const __bf16* arow = A + (size_t)(mbase + ln) * D_;

  for (int kk = 0; kk < D_; kk += 32) {
    v16bf Am = concat8(*(const v8bf*)(arow + kk + hf * 16),
                       *(const v8bf*)(arow + kk + 8 + hf * 16));
#pragma unroll
    for (int nt = 0; nt < 8; ++nt) {
      const __bf16* wrow = WT + (size_t)(nbase + nt * 16 + ln) * D_;
      acc[nt] = wmma_bf16(Am, *(const v16bf*)(wrow + kk + hf * 16), acc[nt]);
    }
  }
#pragma unroll
  for (int nt = 0; nt < 8; ++nt) {
#pragma unroll
    for (int r = 0; r < 8; ++r) {
      out[(size_t)(mbase + r + 8 * hf) * D_ + nbase + nt * 16 + ln] = acc[nt][r];
    }
  }
}

// ---------------------------------------------------------------------------
extern "C" void kernel_launch(void* const* d_in, const int* in_sizes, int n_in,
                              void* d_out, int out_size, void* d_ws, size_t ws_size,
                              hipStream_t stream) {
  const float* x    = (const float*)d_in[0];
  const int*   mask = (const int*)d_in[1];
  const float* bias = (const float*)d_in[2];
  const float* Wqkv = (const float*)d_in[3];
  const float* Wout = (const float*)d_in[4];
  float* out = (float*)d_out;

  char* p = (char*)d_ws;
  __bf16* WqkvT = (__bf16*)p; p += (size_t)3 * D_ * D_ * 2;        // 1.5 MB
  __bf16* WoutT = (__bf16*)p; p += (size_t)D_ * D_ * 2;            // 0.5 MB
  __bf16* Qb    = (__bf16*)p; p += (size_t)B_ * T_ * D_ * 2;       // 8 MB
  __bf16* Kb    = (__bf16*)p; p += (size_t)B_ * T_ * D_ * 2;       // 8 MB
  __bf16* Vt    = (__bf16*)p; p += (size_t)B_ * T_ * D_ * 2;       // 8 MB
  __bf16* Attn  = (__bf16*)p; p += (size_t)B_ * T_ * D_ * 2;       // 8 MB

  cvt_transpose_kernel<<<(3 * D_ * D_ + 255) / 256, 256, 0, stream>>>(
      Wqkv, WqkvT, D_, 3 * D_);
  cvt_transpose_kernel<<<(D_ * D_ + 255) / 256, 256, 0, stream>>>(
      Wout, WoutT, D_, D_);
  qkv_kernel<<<dim3((B_ * T_) / 16, (3 * D_) / 128), 32, 0, stream>>>(
      x, WqkvT, Qb, Kb, Vt);
  attn_kernel<<<dim3(T_ / 32, H_, B_), 32, 0, stream>>>(
      Qb, Kb, Vt, bias, mask, Attn);
  proj_kernel<<<dim3((B_ * T_) / 16, D_ / 128), 32, 0, stream>>>(
      Attn, WoutT, out);
}
